// MHLA_57174604644758
// MI455X (gfx1250) — compile-verified, matrix operand order checked by reading, not run on hardware
//
#include <hip/hip_runtime.h>
#include <hip/hip_bf16.h>

// CDNA5 / gfx1250 — wave32, WMMA 16x16x32 bf16
typedef __attribute__((ext_vector_type(16))) __bf16 v16bf;
typedef __attribute__((ext_vector_type(8)))  float  v8f;

#define B_  2
#define S_  2048
#define D_  1024
#define H_  16
#define HD_ 64
#define M_  (B_ * S_)           // 4096 rows

// 16-bit A/B fragment K-pair swizzle (ISA 7.12.2):
// lanes 0-15 hold K pairs {0..3, 8..11}; lanes 16-31 hold {4..7, 12..15}
__device__ __forceinline__ int pair_idx(int half, int i) {
    return (half ? 4 : 0) + (i < 4 ? i : i + 4);
}

// Fragment from bf16 memory; pair (2k,2k+1) = one dword load.
__device__ __forceinline__ v16bf frag_bf16(const __bf16* base, int ld, int lane) {
    const int r = lane & 15, half = lane >> 4;
    const unsigned int* row = (const unsigned int*)(base + (size_t)r * ld);
    union { v16bf v; unsigned int u[8]; } f;
#pragma unroll
    for (int i = 0; i < 8; ++i) f.u[i] = row[pair_idx(half, i)];
    return f.v;
}

// Fragment from f32 memory with on-the-fly bf16 conversion.
__device__ __forceinline__ v16bf frag_f32(const float* base, int ld, int lane) {
    const int r = lane & 15, half = lane >> 4;
    const float* row = base + (size_t)r * ld;
    v16bf f = {};
#pragma unroll
    for (int i = 0; i < 8; ++i) {
        const int p = pair_idx(half, i);
        f[2 * i]     = (__bf16)row[2 * p];
        f[2 * i + 1] = (__bf16)row[2 * p + 1];
    }
    return f;
}

__device__ __forceinline__ v8f wmma_bf16(v16bf a, v16bf b, v8f c) {
    return __builtin_amdgcn_wmma_f32_16x16x32_bf16(
        false, a, false, b, (short)0, c, false, false);
}

// ---------------------------------------------------------------------------
// Kernel 1: QKV projection.  C[m,n] = sum_k x[m,k] * Wqkv[n,k] + bqkv[n]
// M=4096, N=3072, K=1024.  One wave computes a 32x64 tile (8 WMMAs / K-step,
// A-frags reused 4x, B-frags reused 2x).  The 64-wide N-tile maps to exactly
// one of {q,k,v} and one head -> wave-uniform scatter epilogue.
// q,k stored bf16 (b,h,s,hd); v stored transposed (b,h,hd,s).
// ---------------------------------------------------------------------------
__global__ __launch_bounds__(32) void qkv_gemm(
        const float* __restrict__ x, const float* __restrict__ Wqkv,
        const float* __restrict__ bqkv,
        __bf16* __restrict__ qb, __bf16* __restrict__ kb,
        __bf16* __restrict__ vbT) {
    const int lane = threadIdx.x;
    const int tm = blockIdx.x & 127;   // 128 tiles of 32 rows
    const int tn = blockIdx.x >> 7;    // 48 tiles of 64 cols
    const float* a0b = x + (size_t)(tm * 32) * D_;
    const float* a1b = a0b + (size_t)16 * D_;
    const float* bb  = Wqkv + (size_t)(tn * 64) * D_;
    v8f acc[2][4] = {};
    for (int k = 0; k < D_; k += 32) {
        __builtin_prefetch(a0b + k + 256, 0, 1);   // global_prefetch_b8
        __builtin_prefetch(bb  + k + 256, 0, 1);
        const v16bf a0 = frag_f32(a0b + k, D_, lane);
        const v16bf a1 = frag_f32(a1b + k, D_, lane);
#pragma unroll
        for (int j = 0; j < 4; ++j) {
            const v16bf b = frag_f32(bb + (size_t)(16 * j) * D_ + k, D_, lane);
            acc[0][j] = wmma_bf16(a0, b, acc[0][j]);
            acc[1][j] = wmma_bf16(a1, b, acc[1][j]);
        }
    }
    // D layout: lane<16 -> M=r, N=lane ; lane>=16 -> M=r+8, N=lane-16
    const int n  = lane & 15;
    const int mb = (lane >> 4) * 8;
    const int three = tn >> 4;         // 0=q 1=k 2=v   (wave-uniform)
    const int h = tn & 15;             // head          (wave-uniform)
    const int b_ = (tm * 32) >> 11;    // batch         (32 | 2048)
    const int bh = b_ * H_ + h;
#pragma unroll
    for (int j = 0; j < 4; ++j) {
        const int hd = 16 * j + n;
        const float bias = bqkv[tn * 64 + hd];
#pragma unroll
        for (int ti = 0; ti < 2; ++ti) {
#pragma unroll
            for (int r = 0; r < 8; ++r) {
                const int s = (tm * 32 + 16 * ti + mb + r) & (S_ - 1);
                const __bf16 v = (__bf16)(acc[ti][j][r] + bias);
                if (three == 0)      qb [((size_t)bh * S_ + s) * HD_ + hd] = v;
                else if (three == 1) kb [((size_t)bh * S_ + s) * HD_ + hd] = v;
                else                 vbT[((size_t)bh * HD_ + hd) * S_ + s] = v;
            }
        }
    }
}

// ---------------------------------------------------------------------------
// Kernel 2: causal masked-score attention (no softmax):
//   ctx[t,:] = sum_{s<=t} (q_t . k_s) * v_s     per (b,h)
// One wave per (b,h, 32-row t-pair).  32-wide s-chunks up to the diagonal;
// k/v fragments shared by both 16-row t-tiles (16 WMMAs per chunk).
// Score tiles fully above the diagonal are zeroed, diagonal tiles predicated.
// Scores restaged via LDS (f32) into the A-fragment swizzle.
// ---------------------------------------------------------------------------
__global__ __launch_bounds__(32) void attn(
        const __bf16* __restrict__ qb, const __bf16* __restrict__ kb,
        const __bf16* __restrict__ vbT, float* __restrict__ ctx) {
    __shared__ float sc[32 * 32];
    const int lane = threadIdx.x;
    const int tP = blockIdx.x & 63;        // 64 t-pairs (32 rows each)
    const int bh = blockIdx.x >> 6;        // 32 (b,h)
    const int tBase = tP * 32;
    v16bf qa[2][2];
#pragma unroll
    for (int ti = 0; ti < 2; ++ti)
#pragma unroll
        for (int c = 0; c < 2; ++c)
            qa[ti][c] = frag_bf16(
                qb + ((size_t)bh * S_ + tBase + 16 * ti) * HD_ + 32 * c, HD_, lane);
    v8f acc[2][4] = {};
    const int n  = lane & 15;
    const int mb = (lane >> 4) * 8;
    for (int cch = 0; cch <= tP; ++cch) {
        const int sBase = cch * 32;
        v16bf kf[2][2];
#pragma unroll
        for (int j = 0; j < 2; ++j)
#pragma unroll
            for (int c = 0; c < 2; ++c)
                kf[j][c] = frag_bf16(
                    kb + ((size_t)bh * S_ + sBase + 16 * j) * HD_ + 32 * c, HD_, lane);
#pragma unroll
        for (int ti = 0; ti < 2; ++ti) {
            const int tb = tBase + 16 * ti;
#pragma unroll
            for (int j = 0; j < 2; ++j) {
                const int sT = sBase + 16 * j;
                float* dst = &sc[(16 * ti + mb) * 32 + 16 * j + n];
                if (sT > tb) {             // entirely above diagonal: zeros
#pragma unroll
                    for (int r = 0; r < 8; ++r) dst[r * 32] = 0.f;
                    continue;
                }
                v8f s8 = {};
                s8 = wmma_bf16(qa[ti][0], kf[j][0], s8);
                s8 = wmma_bf16(qa[ti][1], kf[j][1], s8);
                if (sT == tb) {            // diagonal tile: keep s <= t
#pragma unroll
                    for (int r = 0; r < 8; ++r)
                        if (n > mb + r) s8[r] = 0.f;
                }
#pragma unroll
                for (int r = 0; r < 8; ++r) dst[r * 32] = s8[r];
            }
        }
        __syncthreads();
        const v16bf af0 = frag_f32(sc,           32, lane);  // t-tile 0 scores
        const v16bf af1 = frag_f32(sc + 16 * 32, 32, lane);  // t-tile 1 scores
        const __bf16* vrow = vbT + (size_t)bh * HD_ * S_ + sBase;
#pragma unroll
        for (int j4 = 0; j4 < 4; ++j4) {
            const v16bf vf = frag_bf16(vrow + (size_t)(16 * j4) * S_, S_, lane);
            acc[0][j4] = wmma_bf16(af0, vf, acc[0][j4]);
            acc[1][j4] = wmma_bf16(af1, vf, acc[1][j4]);
        }
        __syncthreads();
    }
    const int b_ = bh >> 4, h = bh & 15;
#pragma unroll
    for (int ti = 0; ti < 2; ++ti)
#pragma unroll
        for (int r = 0; r < 8; ++r) {
            const int t = tBase + 16 * ti + mb + r;
            float* orow = ctx + ((size_t)(b_ * S_ + t)) * D_ + h * HD_;
#pragma unroll
            for (int j4 = 0; j4 < 4; ++j4)
                orow[16 * j4 + n] = acc[ti][j4][r];
        }
}

// ---------------------------------------------------------------------------
// Kernel 3: GroupNorm stats per (b,h) over (HD x S): mu and rstd.
// ---------------------------------------------------------------------------
__global__ __launch_bounds__(256) void gn_stats(
        const float* __restrict__ ctx, float* __restrict__ stats) {
    __shared__ float rs[256], rq[256];
    const int bh = blockIdx.x;
    const int b_ = bh >> 4, h = bh & 15;
    const float* base = ctx + (size_t)b_ * S_ * D_ + h * HD_;
    float s = 0.f, q = 0.f;
    for (int i = threadIdx.x; i < S_ * HD_; i += 256) {
        const int t = i >> 6, hd = i & 63;
        const float v = base[(size_t)t * D_ + hd];
        s += v; q += v * v;
    }
    rs[threadIdx.x] = s; rq[threadIdx.x] = q;
    __syncthreads();
    for (int off = 128; off > 0; off >>= 1) {
        if ((int)threadIdx.x < off) {
            rs[threadIdx.x] += rs[threadIdx.x + off];
            rq[threadIdx.x] += rq[threadIdx.x + off];
        }
        __syncthreads();
    }
    if (threadIdx.x == 0) {
        const float inv = 1.f / (float)(S_ * HD_);
        const float mu = rs[0] * inv;
        const float var = rq[0] * inv - mu * mu;
        stats[bh * 2 + 0] = mu;
        stats[bh * 2 + 1] = rsqrtf(var + 1e-5f);
    }
}

// ---------------------------------------------------------------------------
// Kernel 4: y = GN(ctx) @ Wout^T + bout.  32x64 tile per wave; GroupNorm
// affine folded into the A-fragment f32->bf16 conversion.  M=4096, N=K=1024.
// ---------------------------------------------------------------------------
__global__ __launch_bounds__(32) void out_gemm(
        const float* __restrict__ ctx, const float* __restrict__ stats,
        const float* __restrict__ gamma, const float* __restrict__ beta,
        const float* __restrict__ Wout, const float* __restrict__ bout,
        float* __restrict__ y) {
    const int lane = threadIdx.x;
    const int tm = blockIdx.x & 127;   // 128 tiles of 32 rows
    const int tn = blockIdx.x >> 7;    // 16 tiles of 64 cols
    const int r_ = lane & 15, half = lane >> 4;
    const int b_ = (tm * 32) >> 11;    // wave-uniform batch (32 | 2048)
    const float* arow0 = ctx + (size_t)(tm * 32 + r_) * D_;
    const float* arow1 = arow0 + (size_t)16 * D_;
    const float* bb = Wout + (size_t)(tn * 64) * D_;
    v8f acc[2][4] = {};
    for (int k = 0; k < D_; k += 32) {
        __builtin_prefetch(arow0 + k + 256, 0, 1);
        __builtin_prefetch(bb + k + 256, 0, 1);
        const int hh = k >> 6;         // head constant within 32-chunk
        const float mu   = stats[(b_ * H_ + hh) * 2 + 0];
        const float rstd = stats[(b_ * H_ + hh) * 2 + 1];
        v16bf a0 = {}, a1 = {};
#pragma unroll
        for (int i = 0; i < 8; ++i) {
            const int p = pair_idx(half, i);
            const int k0 = k + 2 * p, k1 = k0 + 1;
            const float g0 = rstd * gamma[k0], g1 = rstd * gamma[k1];
            a0[2 * i]     = (__bf16)((arow0[k0] - mu) * g0 + beta[k0]);
            a0[2 * i + 1] = (__bf16)((arow0[k1] - mu) * g1 + beta[k1]);
            a1[2 * i]     = (__bf16)((arow1[k0] - mu) * g0 + beta[k0]);
            a1[2 * i + 1] = (__bf16)((arow1[k1] - mu) * g1 + beta[k1]);
        }
#pragma unroll
        for (int j = 0; j < 4; ++j) {
            const v16bf b = frag_f32(bb + (size_t)(16 * j) * D_ + k, D_, lane);
            acc[0][j] = wmma_bf16(a0, b, acc[0][j]);
            acc[1][j] = wmma_bf16(a1, b, acc[1][j]);
        }
    }
    const int n = lane & 15, mb = (lane >> 4) * 8;
#pragma unroll
    for (int j = 0; j < 4; ++j) {
        const float bias = bout[tn * 64 + 16 * j + n];
#pragma unroll
        for (int ti = 0; ti < 2; ++ti)
#pragma unroll
            for (int r = 0; r < 8; ++r)
                y[(size_t)(tm * 32 + 16 * ti + mb + r) * D_ + tn * 64 + 16 * j + n]
                    = acc[ti][j][r] + bias;
    }
}

// ---------------------------------------------------------------------------
extern "C" void kernel_launch(void* const* d_in, const int* in_sizes, int n_in,
                              void* d_out, int out_size, void* d_ws, size_t ws_size,
                              hipStream_t stream) {
    const float* x    = (const float*)d_in[0];
    const float* Wqkv = (const float*)d_in[1];
    const float* bqkv = (const float*)d_in[2];
    const float* Wout = (const float*)d_in[3];
    const float* bout = (const float*)d_in[4];
    const float* gam  = (const float*)d_in[5];
    const float* bet  = (const float*)d_in[6];
    float* y = (float*)d_out;

    // workspace: q | k | vT (bf16, 8 MiB each) | ctx f32 16 MiB | stats
    char* ws = (char*)d_ws;
    const size_t qkelems = (size_t)B_ * H_ * S_ * HD_;     // 4 Mi elems
    __bf16* qb  = (__bf16*)ws;  ws += qkelems * sizeof(__bf16);
    __bf16* kb  = (__bf16*)ws;  ws += qkelems * sizeof(__bf16);
    __bf16* vbT = (__bf16*)ws;  ws += qkelems * sizeof(__bf16);
    float*  ctx = (float*)ws;   ws += (size_t)M_ * D_ * sizeof(float);
    float*  st  = (float*)ws;

    qkv_gemm<<<dim3(128 * 48), dim3(32), 0, stream>>>(x, Wqkv, bqkv, qb, kb, vbT);
    attn    <<<dim3(32 * 64),  dim3(32), 0, stream>>>(qb, kb, vbT, ctx);
    gn_stats<<<dim3(32),       dim3(256), 0, stream>>>(ctx, st);
    out_gemm<<<dim3(128 * 16), dim3(32), 0, stream>>>(ctx, st, gam, bet, Wout, bout, y);
}
